// EncoderAttn_22179211117207
// MI455X (gfx1250) — matrix operand order repaired and checked by reading, not compiled, not run on hardware
//
#include <hip/hip_runtime.h>
#include <hip/hip_bf16.h>
#include <math.h>

// ---------------------------------------------------------------------------
// Types for CDNA5 WMMA (wave32): D(16x16 f32) = A(16x32 bf16) x B(32x16 bf16) + C
// ---------------------------------------------------------------------------
typedef __attribute__((ext_vector_type(16))) __bf16 v16bf;
typedef __attribute__((ext_vector_type(2)))  __bf16 v2bf;
typedef __attribute__((ext_vector_type(8)))  float  v8f;

union BPack { unsigned int u[8]; v16bf v; };

__device__ __forceinline__ unsigned int pk_bf16(float a, float b) {
#if __has_builtin(__builtin_amdgcn_cvt_pk_bf16_f32)
    v2bf v = __builtin_amdgcn_cvt_pk_bf16_f32(a, b);
    return __builtin_bit_cast(unsigned int, v);
#else
    v2bf v; v.x = (__bf16)a; v.y = (__bf16)b;
    return __builtin_bit_cast(unsigned int, v);
#endif
}

__device__ __forceinline__ v8f wmma_bf16(const BPack& a, const BPack& b, v8f c) {
    return __builtin_amdgcn_wmma_f32_16x16x32_bf16(
        /*neg_a=*/false, a.v, /*neg_b=*/false, b.v,
        /*c_mod=*/(short)0, c, /*reuse_a=*/false, /*reuse_b=*/false);
}

// ---------------------------------------------------------------------------
// GEMM: D[M,N] = act( A[M,K] @ B[K,N] + bias ) + residual
// EXACT tiles: requires M%128==0, N%128==0, K%32==0 (holds for whole net).
// bf16 WMMA, f32 accumulate. Block tile 128x128x32, 8 waves (256 thr).
// ---------------------------------------------------------------------------
#define BM 128
#define BN 128
#define BKK 32

__global__ __launch_bounds__(256) void gemm_bf16_wmma(
    const float* __restrict__ A, const float* __restrict__ B,
    float* __restrict__ D, const float* __restrict__ bias,
    const float* __restrict__ resid, int M, int N, int K, int do_gelu)
{
    __shared__ __align__(16) __bf16 sA[BM * BKK];   // [m][k]
    __shared__ __align__(16) __bf16 sB[BN * BKK];   // [n][k] (transposed)

    const int tid  = threadIdx.x;
    const int wave = tid >> 5;
    const int lane = tid & 31;
    const int half = lane >> 4;
    const int l16  = lane & 15;

    const int m0 = blockIdx.y * BM;
    const int n0 = blockIdx.x * BN;
    const int wm = (wave >> 2) * 64;       // 2 wave rows
    const int wn = (wave & 3) * 32;        // 4 wave cols

    // staging assignments (straight-line, no guards)
    const int arow = tid >> 1;             // 0..127
    const int acol = (tid & 1) * 16;       // 0 or 16
    const int brow = tid >> 3;             // k: 0..31
    const int bcol = (tid & 7) * 16;       // n: 0..112

    const float* agp = A + (size_t)(m0 + arow) * K + acol;
    const float* bgp = B + (size_t)brow * N + n0 + bcol;

    v8f acc[4][2];
    #pragma unroll
    for (int i = 0; i < 4; i++)
        #pragma unroll
        for (int t = 0; t < 2; t++)
            acc[i][t] = (v8f){0.f,0.f,0.f,0.f,0.f,0.f,0.f,0.f};

    for (int k0 = 0; k0 < K; k0 += BKK) {
        // ---- stage A: 4x float4 -> packed bf16 -> 2x b128 LDS stores ----
        {
            const float4* ga = (const float4*)(agp + k0);
            float4 f0 = ga[0], f1 = ga[1], f2 = ga[2], f3 = ga[3];
            uint4 p0, p1;
            p0.x = pk_bf16(f0.x, f0.y); p0.y = pk_bf16(f0.z, f0.w);
            p0.z = pk_bf16(f1.x, f1.y); p0.w = pk_bf16(f1.z, f1.w);
            p1.x = pk_bf16(f2.x, f2.y); p1.y = pk_bf16(f2.z, f2.w);
            p1.z = pk_bf16(f3.x, f3.y); p1.w = pk_bf16(f3.z, f3.w);
            *(uint4*)&sA[arow * BKK + acol]     = p0;
            *(uint4*)&sA[arow * BKK + acol + 8] = p1;
        }
        // ---- stage B (transpose to [n][k]): 4x float4 coalesced loads ----
        {
            const float4* gb = (const float4*)(bgp + (size_t)k0 * N);
            float4 f0 = gb[0], f1 = gb[1], f2 = gb[2], f3 = gb[3];
            float vals[16] = { f0.x,f0.y,f0.z,f0.w, f1.x,f1.y,f1.z,f1.w,
                               f2.x,f2.y,f2.z,f2.w, f3.x,f3.y,f3.z,f3.w };
            #pragma unroll
            for (int i = 0; i < 16; i++)
                sB[(bcol + i) * BKK + brow] = (__bf16)vals[i];
        }
        // prefetch next k-tile while WMMAs run
        if (k0 + BKK < K) {
            __builtin_prefetch(agp + k0 + BKK, 0, 1);
            __builtin_prefetch(bgp + (size_t)(k0 + BKK) * N, 0, 1);
        }
        __syncthreads();

        BPack afr[4], bfr[2];
        #pragma unroll
        for (int i = 0; i < 4; i++) {
            const unsigned int* base =
                (const unsigned int*)&sA[(wm + i * 16 + l16) * BKK];
            #pragma unroll
            for (int j = 0; j < 8; j++) {
                int kp = (j < 4) ? (8 * half + 2 * j)
                                 : (16 + 8 * half + 2 * (j - 4));
                afr[i].u[j] = base[kp >> 1];
            }
        }
        #pragma unroll
        for (int t = 0; t < 2; t++) {
            const unsigned int* base =
                (const unsigned int*)&sB[(wn + t * 16 + l16) * BKK];
            #pragma unroll
            for (int j = 0; j < 8; j++) {
                int kb = (j < 4) ? (8 * half + 2 * j) : (16 + 8 * half + 2 * (j - 4));
                bfr[t].u[j] = base[kb >> 1];
            }
        }
        #pragma unroll
        for (int i = 0; i < 4; i++)
            #pragma unroll
            for (int t = 0; t < 2; t++)
                acc[i][t] = wmma_bf16(afr[i], bfr[t], acc[i][t]);
        __syncthreads();
    }

    // epilogue: bias -> gelu -> residual (exact tiles, no guards)
    #pragma unroll
    for (int i = 0; i < 4; i++)
        #pragma unroll
        for (int t = 0; t < 2; t++)
            #pragma unroll
            for (int r = 0; r < 8; r++) {
                int grow = m0 + wm + i * 16 + r + 8 * half;
                int gcol = n0 + wn + t * 16 + l16;
                float v = acc[i][t][r];
                if (bias) v += bias[gcol];
                if (do_gelu) {
                    float c = v + 0.044715f * v * v * v;
                    v = 0.5f * v * (1.f + tanhf(0.7978845608f * c));
                }
                if (resid) v += resid[(size_t)grow * N + gcol];
                D[(size_t)grow * N + gcol] = v;
            }
}

// ---------------------------------------------------------------------------
// LayerNorm: one wave per row
// ---------------------------------------------------------------------------
__global__ __launch_bounds__(32) void ln_kernel(
    const float* __restrict__ x, float* __restrict__ y,
    const float* __restrict__ g, const float* __restrict__ b, int Dd)
{
    const int row = blockIdx.x;
    const float* xr = x + (size_t)row * Dd;
    float s = 0.f, s2 = 0.f;
    for (int c = threadIdx.x; c < Dd; c += 32) {
        float v = xr[c]; s += v; s2 += v * v;
    }
    #pragma unroll
    for (int off = 16; off; off >>= 1) {
        s  += __shfl_xor(s,  off, 32);
        s2 += __shfl_xor(s2, off, 32);
    }
    float mean = s / (float)Dd;
    float var  = s2 / (float)Dd - mean * mean;
    float rstd = rsqrtf(var + 1e-5f);
    float* yr = y + (size_t)row * Dd;
    for (int c = threadIdx.x; c < Dd; c += 32)
        yr[c] = (xr[c] - mean) * rstd * g[c] + b[c];
}

// ---------------------------------------------------------------------------
// Causal conv stem: x[B,2,L] -> y[B,L,128]  (k=7, left pad 6)
// ---------------------------------------------------------------------------
__global__ void conv_pre_kernel(const float* __restrict__ x,
                                const float* __restrict__ w,
                                const float* __restrict__ bias,
                                float* __restrict__ y, int L)
{
    int idx = blockIdx.x * blockDim.x + threadIdx.x;
    int total = 2 * L * 128;
    if (idx >= total) return;
    int c = idx & 127;
    int l = (idx >> 7) % L;
    int b = idx / (128 * L);
    float acc = bias[c];
    #pragma unroll
    for (int i = 0; i < 2; i++)
        #pragma unroll
        for (int k = 0; k < 7; k++) {
            int p = l - 6 + k;
            if (p >= 0)
                acc += x[((size_t)b * 2 + i) * L + p] * w[(c * 2 + i) * 7 + k];
        }
    y[idx] = acc;   // idx == (b*L + l)*128 + c
}

// ---------------------------------------------------------------------------
// Causal conv head: x[B*L, C] -> y[B*L, 64]  (k=3, left pad 2), w[64,C,3]
// ---------------------------------------------------------------------------
__global__ void conv_post_kernel(const float* __restrict__ x,
                                 const float* __restrict__ w,
                                 const float* __restrict__ bias,
                                 float* __restrict__ y, int L, int C)
{
    int idx = blockIdx.x * blockDim.x + threadIdx.x;
    int total = 2 * L * 64;
    if (idx >= total) return;
    int o = idx & 63;
    int l = (idx >> 6) % L;
    int b = idx / (64 * L);
    float acc = bias[o];
    #pragma unroll
    for (int k = 0; k < 3; k++) {
        int p = l - 2 + k;
        if (p < 0) continue;
        const float* xr = x + ((size_t)b * L + p) * C;
        const float* wr = w + (size_t)o * C * 3 + k;
        for (int c = 0; c < C; c++) acc += xr[c] * wr[c * 3];
    }
    y[((size_t)b * L + l) * 64 + o] = acc;
}

// ---------------------------------------------------------------------------
// Local windowed causal flash attention (W=256, look-back 1 window).
// qkv[M,1536] (q|k|v, head h at [h*64,(h+1)*64)); out[M,512].
// Block = (batch, head, window); 16 waves, one 16-query tile per wave.
// ---------------------------------------------------------------------------
#define AWIN 256
#define AHEADS 8

__global__ __launch_bounds__(512) void attn_kernel(
    const float* __restrict__ qkv, float* __restrict__ out, int n)
{
    __shared__ __align__(16) __bf16 sP[16 * 512];  // 16 waves x (16x32 P tile)

    const int tid  = threadIdx.x;
    const int wave = tid >> 5;
    const int lane = tid & 31;
    const int half = lane >> 4;
    const int l16  = lane & 15;

    const int nw   = n / AWIN;
    const int win  = blockIdx.x % nw;
    const int head = (blockIdx.x / nw) % AHEADS;
    const int bb   = blockIdx.x / (nw * AHEADS);
    const int qt   = wave;

    const size_t tok0  = (size_t)bb * n + (size_t)win * AWIN;
    const float* kbase = qkv + (size_t)bb * n * 1536 + 512  + head * 64;
    const float* vbase = qkv + (size_t)bb * n * 1536 + 1024 + head * 64;

    // ---- load Q fragments (pre-scaled by 1/sqrt(64)); float2 pair loads ----
    BPack aq[2];
    {
        const size_t qrow = tok0 + (size_t)qt * 16 + l16;
        const float* qp = qkv + qrow * 1536 + head * 64;
        #pragma unroll
        for (int kk = 0; kk < 2; kk++)
            #pragma unroll
            for (int j = 0; j < 8; j++) {
                int d0 = kk * 32 + ((j < 4) ? (8 * half + 2 * j)
                                            : (16 + 8 * half + 2 * (j - 4)));
                float2 f = *(const float2*)(qp + d0);
                aq[kk].u[j] = pk_bf16(f.x * 0.125f, f.y * 0.125f);
            }
    }

    v8f o[4];
    #pragma unroll
    for (int t = 0; t < 4; t++) o[t] = (v8f){0.f,0.f,0.f,0.f,0.f,0.f,0.f,0.f};
    float mr[8], lr[8];
    #pragma unroll
    for (int r = 0; r < 8; r++) { mr[r] = -1e30f; lr[r] = 0.f; }

    __bf16* pbase = sP + wave * 512;

    for (int kc = 0; kc < 16; kc++) {          // 16 chunks of 32 keys (512 total)
        // ---- S = Q K^T (two 16x16 tiles); clamped ptr + select, float2 ----
        v8f s[2];
        #pragma unroll
        for (int nt = 0; nt < 2; nt++) {
            int key = kc * 32 + nt * 16 + l16;             // 0..511
            long keypos = (long)win * AWIN + key - AWIN;   // abs seq pos
            bool valid = keypos >= 0;
            const float* kp = kbase + (valid ? keypos : 0) * 1536;
            BPack bk[2];
            #pragma unroll
            for (int kk = 0; kk < 2; kk++)
                #pragma unroll
                for (int j = 0; j < 8; j++) {
                    int d0 = kk * 32 + ((j < 4) ? (8 * half + 2 * j) : (16 + 8 * half + 2 * (j - 4)));
                    float2 f = *(const float2*)(kp + d0);
                    float a = valid ? f.x : 0.f;
                    float b = valid ? f.y : 0.f;
                    bk[kk].u[j] = pk_bf16(a, b);
                }
            v8f z = (v8f){0.f,0.f,0.f,0.f,0.f,0.f,0.f,0.f};
            z = wmma_bf16(aq[0], bk[0], z);
            z = wmma_bf16(aq[1], bk[1], z);
            s[nt] = z;
        }

        // ---- mask + online softmax update ----
        float pv[2][8], alpha[8];
        #pragma unroll
        for (int r = 0; r < 8; r++) {
            int Mrow = r + 8 * half;
            int qi = qt * 16 + Mrow;
            float s0 = s[0][r], s1 = s[1][r];
            {
                int k0 = kc * 32 + l16;       int kj0 = k0 - AWIN;
                int k1 = kc * 32 + 16 + l16;  int kj1 = k1 - AWIN;
                if ((kj0 > qi) || ((long)win * AWIN + kj0 < 0)) s0 = -1e30f;
                if ((kj1 > qi) || ((long)win * AWIN + kj1 < 0)) s1 = -1e30f;
            }
            float mx = fmaxf(s0, s1);
            #pragma unroll
            for (int off = 1; off < 16; off <<= 1)
                mx = fmaxf(mx, __shfl_xor(mx, off, 32));
            float mnew = fmaxf(mr[r], mx);
            float al = __expf(mr[r] - mnew);
            float p0 = __expf(s0 - mnew), p1 = __expf(s1 - mnew);
            float rs = p0 + p1;
            #pragma unroll
            for (int off = 1; off < 16; off <<= 1)
                rs += __shfl_xor(rs, off, 32);
            lr[r] = lr[r] * al + rs;
            mr[r] = mnew;
            alpha[r] = al;
            pv[0][r] = p0; pv[1][r] = p1;
        }
        #pragma unroll
        for (int t = 0; t < 4; t++)
            #pragma unroll
            for (int r = 0; r < 8; r++) o[t][r] *= alpha[r];

        // ---- P (D-layout) -> LDS -> A-layout fragment ----
        #pragma unroll
        for (int nt = 0; nt < 2; nt++)
            #pragma unroll
            for (int r = 0; r < 8; r++) {
                int Mrow = r + 8 * half;
                pbase[Mrow * 32 + nt * 16 + l16] = (__bf16)pv[nt][r];
            }
        asm volatile("s_wait_dscnt 0" ::: "memory");   // wave-local LDS RAW

        BPack ap;
        #pragma unroll
        for (int j = 0; j < 8; j++) {
            int kp = (j < 4) ? (8 * half + 2 * j)
                             : (16 + 8 * half + 2 * (j - 4));
            ap.u[j] = *(const unsigned int*)&pbase[l16 * 32 + kp];
        }

        // ---- O += P @ V (clamped ptr + select, branch-free) ----
        #pragma unroll
        for (int t = 0; t < 4; t++) {
            int dcol = t * 16 + l16;
            BPack bv;
            #pragma unroll
            for (int j = 0; j < 8; j++) {
                int kv = (j < 4) ? (8 * half + 2 * j) : (16 + 8 * half + 2 * (j - 4)); // even key in chunk
                long keypos = (long)win * AWIN + kc * 32 + kv - AWIN;
                bool valid = keypos >= 0;                   // pair shares validity
                const float* vp = vbase + (valid ? keypos : 0) * 1536 + dcol;
                float a = valid ? vp[0]    : 0.f;
                float b = valid ? vp[1536] : 0.f;           // keypos+1 row
                bv.u[j] = pk_bf16(a, b);
            }
            o[t] = wmma_bf16(ap, bv, o[t]);
        }
    }

    // ---- normalize + store ----
    #pragma unroll
    for (int t = 0; t < 4; t++)
        #pragma unroll
        for (int r = 0; r < 8; r++) {
            int Mrow = r + 8 * half;
            size_t row = tok0 + (size_t)qt * 16 + Mrow;
            out[row * 512 + head * 64 + t * 16 + l16] = o[t][r] / lr[r];
        }
}

// ---------------------------------------------------------------------------
// Host-side pipeline
// ---------------------------------------------------------------------------
static inline void launch_gemm(const float* A, const float* B, float* D,
                               const float* bias, const float* resid,
                               int M, int N, int K, int gelu, hipStream_t s)
{
    dim3 grid(N / BN, M / BM);
    gemm_bf16_wmma<<<grid, 256, 0, s>>>(A, B, D, bias, resid, M, N, K, gelu);
}

extern "C" void kernel_launch(void* const* d_in, const int* in_sizes, int n_in,
                              void* d_out, int out_size, void* d_ws, size_t ws_size,
                              hipStream_t stream)
{
    (void)in_sizes; (void)n_in; (void)out_size; (void)ws_size;
    int idx = 0;
    auto nxt = [&]() { return (const float*)d_in[idx++]; };

    const float* x_in  = nxt();
    const float* pre_w = nxt();
    const float* pre_b = nxt();

    struct LayerP { const float *ag,*ab,*wqkv,*wo,*fg,*fb,*w1,*b1,*w2,*b2; };
    struct StageP { const float *dw,*db; LayerP lay[4]; int depth; };
    const int depths[4] = {2, 2, 2, 4};
    StageP st[4];
    for (int i = 0; i < 4; i++) {
        st[i].dw = nxt(); st[i].db = nxt(); st[i].depth = depths[i];
        for (int l = 0; l < depths[i]; l++) {
            LayerP& L = st[i].lay[l];
            L.ag = nxt(); L.ab = nxt(); L.wqkv = nxt(); L.wo = nxt();
            L.fg = nxt(); L.fb = nxt();
            L.w1 = nxt(); L.b1 = nxt(); L.w2 = nxt(); L.b2 = nxt();
        }
    }
    const float* post_w = nxt();
    const float* post_b = nxt();
    const float* norm_g = nxt();
    const float* norm_b = nxt();

    char* ws = (char*)d_ws;
    const size_t MB = 1024ull * 1024ull;
    float* X0 = (float*)(ws);
    float* X1 = (float*)(ws + 34 * MB);
    float* H  = (float*)(ws + 68 * MB);
    float* QV = (float*)(ws + 102 * MB);   // qkv / ff hidden (largest)
    float* AO = (float*)(ws + 204 * MB);

    const int B = 2, L0 = 16384;

    // stem conv -> X0 [B*L0, 128]
    {
        int total = B * L0 * 128;
        conv_pre_kernel<<<(total + 255) / 256, 256, 0, stream>>>(
            x_in, pre_w, pre_b, X0, L0);
    }

    float* Xc = X0; float* Xo = X1;
    int l = L0, c = 128;
    const int douts[4] = {128, 256, 512, 1024};

    for (int i = 0; i < 4; i++) {
        const int dout = douts[i];
        const int lnew = l / 2;
        const int M = B * lnew;
        // downsample: reinterpret [b,l,c] as [M, 2c], GEMM to [M, dout]
        launch_gemm(Xc, st[i].dw, Xo, st[i].db, nullptr, M, dout, 2 * c, 0, stream);
        { float* t = Xc; Xc = Xo; Xo = t; }
        l = lnew; c = dout;

        for (int li = 0; li < st[i].depth; li++) {
            const LayerP& P = st[i].lay[li];
            // --- local attention block ---
            ln_kernel<<<M, 32, 0, stream>>>(Xc, H, P.ag, P.ab, c);
            launch_gemm(H, P.wqkv, QV, nullptr, nullptr, M, 1536, c, 0, stream);
            int nwin = l / AWIN;
            attn_kernel<<<B * AHEADS * nwin, 512, 0, stream>>>(QV, AO, l);
            launch_gemm(AO, P.wo, Xc, nullptr, Xc, M, c, 512, 0, stream); // +resid
            // --- feed-forward block ---
            ln_kernel<<<M, 32, 0, stream>>>(Xc, H, P.fg, P.fb, c);
            launch_gemm(H,  P.w1, QV, P.b1, nullptr, M, 4 * c, c, 1, stream); // gelu
            launch_gemm(QV, P.w2, Xc, P.b2, Xc,      M, c, 4 * c, 0, stream); // +resid
        }
    }

    // head conv (1024 -> 64, k=3) then final LN -> d_out
    {
        int total = B * l * 64;
        conv_post_kernel<<<(total + 255) / 256, 256, 0, stream>>>(
            Xc, post_w, post_b, H, l, c);
        ln_kernel<<<B * l, 32, 0, stream>>>(H, (float*)d_out, norm_g, norm_b, 64);
    }
}
